// RelativeMultiHeadAttention_55594056680055
// MI455X (gfx1250) — matrix operand order repaired and checked by reading, not compiled, not run on hardware
//
#include <hip/hip_runtime.h>
#include <hip/hip_bf16.h>

// Shapes
#define BB   2
#define SS   2048
#define HID_ 512
#define NH_  8
#define DD   64

typedef __attribute__((ext_vector_type(16))) __bf16 v16bf;
typedef __attribute__((ext_vector_type(8)))  __bf16 v8bf;
typedef __attribute__((ext_vector_type(8)))  float  v8f;
typedef __attribute__((ext_vector_type(16))) float  v16f;

static __device__ inline v16bf cvt16_to_bf(v16f x) {
    v16bf r;
#pragma unroll
    for (int i = 0; i < 16; i++) r[i] = (__bf16)x[i];
    return r;
}

// ---------------------------------------------------------------------------
// Kernel 1: fused projection GEMM.  out = X @ W^T + bias, per-mode epilogue:
//   mode 0: Q  -> bf16 [b,h,s,d]
//   mode 1: K' = scale[h]*K + rel[j,h,d] -> bf16 [b,h,s,d]  (bias+scale+rel folded)
//   mode 2: V  -> bf16 [b,h,d,s]  (transposed for PV B-operand)
//   mode 3: RV -> f32  [b,s,hid]
// One wave computes a 16x16 tile via v_wmma_f32_16x16x32_bf16 over K=512,
// with two independent accumulators to break the WMMA->WMMA RAW chain.
// ---------------------------------------------------------------------------
__global__ __launch_bounds__(256) void proj_kernel(
    const float* __restrict__ query, const float* __restrict__ key,
    const float* __restrict__ value,
    const float* __restrict__ Wq, const float* __restrict__ bq,
    const float* __restrict__ Wk, const float* __restrict__ bk,
    const float* __restrict__ Wv, const float* __restrict__ bv,
    const float* __restrict__ Wrv, const float* __restrict__ brv,
    const float* __restrict__ rel, const float* __restrict__ scaling,
    __bf16* __restrict__ Qbf, __bf16* __restrict__ Kbf,
    __bf16* __restrict__ Vt, float* __restrict__ RVf)
{
    const int mode = blockIdx.z;
    const float* X; const float* W; const float* bias;
    if      (mode == 0) { X = query; W = Wq;  bias = bq;  }
    else if (mode == 1) { X = key;   W = Wk;  bias = bk;  }
    else if (mode == 2) { X = value; W = Wv;  bias = bv;  }
    else                { X = value; W = Wrv; bias = brv; }

    const int lane = threadIdx.x & 31;
    const int wid  = threadIdx.x >> 5;
    const int cc   = lane & 15;   // B/C column, A row
    const int hf   = lane >> 4;   // lane half

    const int n0 = (blockIdx.y * 8 + wid) * 16;   // token-tile base
    const int m0 = blockIdx.x * 16;               // feature-tile base

    const int arow = n0 + cc;     // A-matrix row (token)
    const int wrow = m0 + cc;     // W row supplying B column cc

    v8f acc0, acc1;
#pragma unroll
    for (int r = 0; r < 8; r++) { acc0[r] = 0.0f; acc1[r] = 0.0f; }

#pragma unroll 4
    for (int k0 = 0; k0 < HID_; k0 += 64) {
#pragma unroll
        for (int half_step = 0; half_step < 2; half_step++) {
            const int kk = k0 + half_step * 32;
            // A operand: row arow; K elems hf*16 .. +16
            v8f x1 = *(const v8f*)(X + arow * HID_ + kk + hf * 16);
            v8f x2 = *(const v8f*)(X + arow * HID_ + kk + hf * 16 + 8);
            v16bf a;
#pragma unroll
            for (int i = 0; i < 8; i++) { a[i] = (__bf16)x1[i]; a[i + 8] = (__bf16)x2[i]; }
            // B operand: col cc, K elems hf*16 .. +16 (B[k][n] = W[m0+n][k])
            v16f wv = *(const v16f*)(W + wrow * HID_ + kk + hf * 16);
            v16bf bmat = cvt16_to_bf(wv);
            if (half_step == 0)
                acc0 = __builtin_amdgcn_wmma_f32_16x16x32_bf16(
                    false, a, false, bmat, (short)0, acc0, false, false);
            else
                acc1 = __builtin_amdgcn_wmma_f32_16x16x32_bf16(
                    false, a, false, bmat, (short)0, acc1, false, false);
        }
    }
    v8f acc = acc0 + acc1;

    const int m  = m0 + cc;           // output feature (col)
    const float bia = bias[m];
    const int h = m >> 6, d = m & 63;
    const float sc = (mode == 1) ? scaling[h] : 0.0f;

#pragma unroll
    for (int r = 0; r < 8; r++) {
        const int n = n0 + r + 8 * hf;        // C-layout row
        const int bidx = n >> 11, s = n & (SS - 1);
        float val = acc[r] + bia;
        if (mode == 0) {
            Qbf[(((bidx * NH_) + h) * SS + s) * DD + d] = (__bf16)val;
        } else if (mode == 1) {
            float v2 = sc * val + rel[(s * NH_ + h) * DD + d];
            Kbf[(((bidx * NH_) + h) * SS + s) * DD + d] = (__bf16)v2;
        } else if (mode == 2) {
            Vt[(((bidx * NH_) + h) * DD + d) * SS + s] = (__bf16)val;
        } else {
            RVf[(size_t)n * HID_ + m] = val;
        }
    }
}

// ---------------------------------------------------------------------------
// Kernel 2: flash attention.  One wave owns 16 query rows; loops over key
// tiles of 64 with online softmax.  scores = Q @ K'^T (scale+rel pre-folded).
// P re-laid C->A through padded LDS; PV uses transposed V for contiguous B.
// Next j-tile is prefetched (global_prefetch_b8) while computing current.
// out = O/l + RV.
// ---------------------------------------------------------------------------
__global__ __launch_bounds__(256) void attn_kernel(
    const __bf16* __restrict__ Qbf, const __bf16* __restrict__ Kbf,
    const __bf16* __restrict__ Vt, const float* __restrict__ RVf,
    float* __restrict__ out)
{
    __shared__ __align__(16) __bf16 plds[8][16 * 80];   // stride 80 halves: 16B aligned chunks

    const int lane = threadIdx.x & 31;
    const int wid  = threadIdx.x >> 5;
    const int cc   = lane & 15;
    const int hf   = lane >> 4;
    const int bidx = blockIdx.z, h = blockIdx.y;
    const int i0   = blockIdx.x * 128 + wid * 16;

    const __bf16* Qb = Qbf + (size_t)((bidx * NH_ + h) * SS) * DD;
    const __bf16* Kb = Kbf + (size_t)((bidx * NH_ + h) * SS) * DD;
    const __bf16* Vb = Vt  + (size_t)((bidx * NH_ + h) * DD) * SS;

    // Preload Q in A layout: 2 chunks of 16x32
    v16bf qa[2];
#pragma unroll
    for (int kc = 0; kc < 2; kc++) {
        const __bf16* p = Qb + (i0 + cc) * DD + kc * 32;
        v8bf lo = *(const v8bf*)(p + hf * 16);
        v8bf hi = *(const v8bf*)(p + hf * 16 + 8);
        qa[kc] = __builtin_shufflevector(lo, hi,
            0, 1, 2, 3, 4, 5, 6, 7, 8, 9, 10, 11, 12, 13, 14, 15);
    }

    float m_[8], l_[8];
    v8f o[4];
#pragma unroll
    for (int r = 0; r < 8; r++) { m_[r] = -1e30f; l_[r] = 0.0f; }
#pragma unroll
    for (int nt = 0; nt < 4; nt++)
#pragma unroll
        for (int r = 0; r < 8; r++) o[nt][r] = 0.0f;

    for (int j0 = 0; j0 < SS; j0 += 64) {
        // ---- prefetch next j-tile (K rows: 64 x 128B; V rows: 64 x 128B)
        if (j0 + 64 < SS) {
            const int jn = j0 + 64;
            __builtin_prefetch(Kb + (jn + lane) * DD, 0, 3);
            __builtin_prefetch(Kb + (jn + 32 + lane) * DD, 0, 3);
            __builtin_prefetch(Vb + lane * SS + jn, 0, 3);
            __builtin_prefetch(Vb + (32 + lane) * SS + jn, 0, 3);
        }
        // ---- scores 16x64 = Q @ K'^T  (independent accumulators per kc)
        v8f sc[4];
#pragma unroll
        for (int nt = 0; nt < 4; nt++) {
            const int keyrow = j0 + nt * 16 + cc;   // B col = key index
            v16bf kb0 = *(const v16bf*)(Kb + keyrow * DD + 0  + hf * 16);
            v16bf kb1 = *(const v16bf*)(Kb + keyrow * DD + 32 + hf * 16);
            v8f z;
#pragma unroll
            for (int r = 0; r < 8; r++) z[r] = 0.0f;
            v8f c0 = __builtin_amdgcn_wmma_f32_16x16x32_bf16(
                false, qa[0], false, kb0, (short)0, z, false, false);
            v8f c1 = __builtin_amdgcn_wmma_f32_16x16x32_bf16(
                false, qa[1], false, kb1, (short)0, z, false, false);
            sc[nt] = c0 + c1;
        }
        // ---- online softmax stats (row = r + 8*hf, spread over 16 lanes)
        float rmax[8];
#pragma unroll
        for (int r = 0; r < 8; r++)
            rmax[r] = fmaxf(fmaxf(sc[0][r], sc[1][r]), fmaxf(sc[2][r], sc[3][r]));
#pragma unroll
        for (int mask = 1; mask < 16; mask <<= 1)
#pragma unroll
            for (int r = 0; r < 8; r++)
                rmax[r] = fmaxf(rmax[r], __shfl_xor(rmax[r], mask, 32));
        float alpha[8];
#pragma unroll
        for (int r = 0; r < 8; r++) {
            float mn = fmaxf(m_[r], rmax[r]);
            alpha[r] = __expf(m_[r] - mn);
            m_[r] = mn;
        }
        float rsum[8];
#pragma unroll
        for (int r = 0; r < 8; r++) rsum[r] = 0.0f;
#pragma unroll
        for (int nt = 0; nt < 4; nt++)
#pragma unroll
            for (int r = 0; r < 8; r++) {
                float p = __expf(sc[nt][r] - m_[r]);
                sc[nt][r] = p;
                rsum[r] += p;
            }
#pragma unroll
        for (int mask = 1; mask < 16; mask <<= 1)
#pragma unroll
            for (int r = 0; r < 8; r++)
                rsum[r] += __shfl_xor(rsum[r], mask, 32);
#pragma unroll
        for (int r = 0; r < 8; r++) l_[r] = l_[r] * alpha[r] + rsum[r];
#pragma unroll
        for (int nt = 0; nt < 4; nt++)
#pragma unroll
            for (int r = 0; r < 8; r++) o[nt][r] *= alpha[r];

        // ---- P: C layout -> LDS (bf16) -> A layout
#pragma unroll
        for (int nt = 0; nt < 4; nt++)
#pragma unroll
            for (int r = 0; r < 8; r++)
                plds[wid][(r + 8 * hf) * 80 + nt * 16 + cc] = (__bf16)sc[nt][r];

        v16bf pa[2];
#pragma unroll
        for (int kc = 0; kc < 2; kc++) {
            const __bf16* p = &plds[wid][cc * 80 + kc * 32];
            v8bf lo = *(const v8bf*)(p + hf * 16);
            v8bf hi = *(const v8bf*)(p + hf * 16 + 8);
            pa[kc] = __builtin_shufflevector(lo, hi,
                0, 1, 2, 3, 4, 5, 6, 7, 8, 9, 10, 11, 12, 13, 14, 15);
        }
        // ---- O += P @ V   (B operand from transposed V: contiguous in s)
#pragma unroll
        for (int nt = 0; nt < 4; nt++) {
#pragma unroll
            for (int kc = 0; kc < 2; kc++) {
                v16bf vb = *(const v16bf*)(Vb + (nt * 16 + cc) * SS +
                                           j0 + kc * 32 + hf * 16);
                o[nt] = __builtin_amdgcn_wmma_f32_16x16x32_bf16(
                    false, pa[kc], false, vb, (short)0, o[nt], false, false);
            }
        }
    }

    // ---- epilogue: out = O/l + RV
#pragma unroll
    for (int r = 0; r < 8; r++) {
        const float inv = 1.0f / l_[r];
        const int i = i0 + r + 8 * hf;
#pragma unroll
        for (int nt = 0; nt < 4; nt++) {
            const int col = h * DD + nt * 16 + cc;
            const size_t idx = (size_t)(bidx * SS + i) * HID_ + col;
            out[idx] = o[nt][r] * inv + RVf[idx];
        }
    }
}

// ---------------------------------------------------------------------------
extern "C" void kernel_launch(void* const* d_in, const int* in_sizes, int n_in,
                              void* d_out, int out_size, void* d_ws, size_t ws_size,
                              hipStream_t stream) {
    const float* query   = (const float*)d_in[0];
    const float* key     = (const float*)d_in[1];
    const float* value   = (const float*)d_in[2];
    const float* Wq      = (const float*)d_in[3];
    const float* bq      = (const float*)d_in[4];
    const float* Wk      = (const float*)d_in[5];
    const float* bk      = (const float*)d_in[6];
    const float* Wv      = (const float*)d_in[7];
    const float* bv      = (const float*)d_in[8];
    const float* Wrv     = (const float*)d_in[9];
    const float* brv     = (const float*)d_in[10];
    const float* rel     = (const float*)d_in[11];
    const float* scaling = (const float*)d_in[12];
    // d_in[13] = relative_position (== S for shape-consistent reference)

    char* ws = (char*)d_ws;
    __bf16* Qbf = (__bf16*)(ws);                       // 4 MB
    __bf16* Kbf = (__bf16*)(ws + (4u  << 20));         // 4 MB  (scale+rel folded)
    __bf16* Vt  = (__bf16*)(ws + (8u  << 20));         // 4 MB  (transposed)
    float*  RVf = (float*) (ws + (12u << 20));         // 8 MB

    dim3 gproj(HID_ / 16, (BB * SS) / 128, 4);         // 32 x 32 x 4
    proj_kernel<<<gproj, 256, 0, stream>>>(query, key, value,
        Wq, bq, Wk, bk, Wv, bv, Wrv, brv, rel, scaling,
        Qbf, Kbf, Vt, RVf);

    dim3 gattn(SS / 128, NH_, BB);                     // 16 x 8 x 2
    attn_kernel<<<gattn, 256, 0, stream>>>(Qbf, Kbf, Vt, RVf, (float*)d_out);
}
